// SparseConvolution2D_75625784148187
// MI455X (gfx1250) — compile-verified, matrix operand order checked by reading, not run on hardware
//
#include <hip/hip_runtime.h>

// ---------------------------------------------------------------------------
// Problem geometry (from reference):
//   inputs (8,64,64,256) f32 ; P (256,256) ; Q (256,3,3,16) ; S (256,16,256)
//   O (8,62,62,256) f32
// Folded:  O = conv3x3_valid(inputs, WW) with WW[dy,dx,c,j] = sum_i P[i,c]*W1[tap,i,j]
//          W1[tap,i,j] = sum_k Q[i,dy,dx,k]*S[i,k,j]
// Main conv: 36.3 GFLOP on v_wmma_f32_16x16x32_bf16; inputs (16.7MB bf16) and
// folded weights (1.2MB) are L2-resident; HBM traffic ~65MB (~2.8us floor).
// ---------------------------------------------------------------------------

typedef __attribute__((ext_vector_type(16))) __bf16       v16bf;
typedef __attribute__((ext_vector_type(8)))  float        v8f;
typedef __attribute__((ext_vector_type(4)))  unsigned int v4u;

#define B_    8
#define H_    64
#define W_    64
#define C_    256
#define HO_   62
#define WO_   62
#define F_    256
#define BAS_  16
#define TAPS_ 9
#define M_    (B_ * HO_ * WO_)   // 30752 = 961 * 32 exactly
#define KB_   8                  // k-blocks of 32 per tap (C=256)
#define NT_   16                 // n-tiles of 16 (F=256)

// workspace layout (bytes)
#define WS_INBF_OFF  ((size_t)0)
#define WS_INBF_SZ   ((size_t)B_ * H_ * W_ * C_ * 2)        // 16,777,216
#define WS_W1_OFF    (WS_INBF_OFF + WS_INBF_SZ)
#define WS_W1_SZ     ((size_t)TAPS_ * C_ * F_ * 4)          //  2,359,296
#define WS_FRAG_OFF  (WS_W1_OFF + WS_W1_SZ)
#define WS_FRAG_SZ   ((size_t)TAPS_ * KB_ * NT_ * 512 * 2)  //  1,179,648

static __device__ __forceinline__ unsigned short f32_to_bf16(float f) {
  unsigned int u = __float_as_uint(f);
  unsigned int r = (u + 0x7FFFu + ((u >> 16) & 1u)) >> 16;  // round-nearest-even
  return (unsigned short)r;
}

// ---- Stage 0: inputs f32 -> bf16 (L2-resident staging buffer) --------------
__global__ void cvt_bf16_kernel(const float* __restrict__ in,
                                unsigned short* __restrict__ out, int n) {
  int i = blockIdx.x * blockDim.x + threadIdx.x;
  if (i < n) out[i] = f32_to_bf16(in[i]);
}

// ---- Stage 1: W1[tap,i,j] = sum_k Q[i,tap,k] * S[i,k,j]  (f32) -------------
__global__ void w1_kernel(const float* __restrict__ Q, const float* __restrict__ S,
                          float* __restrict__ w1) {
  int idx = blockIdx.x * blockDim.x + threadIdx.x;
  if (idx >= TAPS_ * C_ * F_) return;
  int j   = idx & (F_ - 1);
  int i   = (idx >> 8) & (C_ - 1);
  int tap = idx >> 16;
  const float* q = Q + ((size_t)i * TAPS_ + tap) * BAS_;      // Q[i][dy][dx][*]
  const float* s = S + (size_t)i * BAS_ * F_ + j;             // S[i][*][j]
  float acc = 0.f;
#pragma unroll
  for (int k = 0; k < BAS_; ++k) acc += q[k] * s[(size_t)k * F_];
  w1[((size_t)tap * C_ + i) * F_ + j] = acc;
}

// ---- Stage 2: fold P, emit WW directly in WMMA B-fragment layout -----------
// frag[tap][kb][nt][lane][h]:  lane = n_local + 16*khalf,
//   halfs 0..7  -> k_local = khalf*8 + h
//   halfs 8..15 -> k_local = 16 + khalf*8 + (h-8)
__global__ void frag_kernel(const float* __restrict__ P, const float* __restrict__ w1,
                            unsigned short* __restrict__ frag) {
  int idx = blockIdx.x * blockDim.x + threadIdx.x;
  if (idx >= TAPS_ * KB_ * NT_ * 32 * 16) return;
  int h    = idx & 15;
  int lane = (idx >> 4) & 31;
  int nt   = (idx >> 9) & 15;
  int kb   = (idx >> 13) & 7;
  int tap  = idx >> 16;
  int khalf   = lane >> 4;
  int n_local = lane & 15;
  int k_local = ((h & 8) << 1) + khalf * 8 + (h & 7);
  int c = kb * 32 + k_local;
  int j = nt * 16 + n_local;
  const float* w = w1 + (size_t)tap * C_ * F_ + j;            // W1[tap][*][j]
  const float* p = P + c;                                     // P[*][c]
  float acc = 0.f;
  for (int i = 0; i < C_; ++i) acc += p[(size_t)i * C_] * w[(size_t)i * F_];
  frag[(((size_t)(tap * KB_ + kb) * NT_ + nt) * 512) + lane * 16 + h] = f32_to_bf16(acc);
}

// ---- Stage 3: main conv as implicit GEMM on v_wmma_f32_16x16x32_bf16 -------
// 4 waves/WG; WG owns 32 pixels (2 M-tiles) x all 256 filters.
// Each wave: 2 M-tiles x 4 N-tiles = 8 accumulators; every B fragment feeds
// two WMMAs (halves B traffic); kb loop fully unrolled -> 64 WMMAs per tap.
__global__ __launch_bounds__(128) void conv_wmma_kernel(
    const unsigned short* __restrict__ inbf,
    const unsigned short* __restrict__ frag,
    float* __restrict__ out) {
  const int lane  = threadIdx.x & 31;
  const int wave  = threadIdx.x >> 5;
  const int mbase = blockIdx.x * 32;

  // A-load rows for this lane (WMMA A layout: lanes L and L+16 share row M=L&15)
  const unsigned short* apix[2];
#pragma unroll
  for (int mt = 0; mt < 2; ++mt) {
    const int ml  = mbase + mt * 16 + (lane & 15);
    const int b   = ml / (HO_ * WO_);
    const int rem = ml - b * (HO_ * WO_);
    const int oy  = rem / WO_;
    const int ox  = rem - oy * WO_;
    apix[mt] = inbf + (size_t)((b * H_ + oy) * W_ + ox) * C_;
  }
  const int khalf8 = (lane >> 4) * 8;  // lanes 16..31 take the odd k-octets

  v8f acc[2][4] = {};

  for (int tap = 0; tap < TAPS_; ++tap) {
    const int dy = tap / 3;
    const int dx = tap - dy * 3;
    const int toff = (dy * W_ + dx) * C_;
    const unsigned short* arow0 = apix[0] + toff;
    const unsigned short* arow1 = apix[1] + toff;
    if (tap + 1 < TAPS_) {  // gfx1250 global_prefetch of next tap's input rows
      const int ndy = (tap + 1) / 3, ndx = (tap + 1) - ndy * 3;
      const int noff = (ndy * W_ + ndx) * C_;
      __builtin_prefetch(apix[0] + noff, 0, 1);
      __builtin_prefetch(apix[1] + noff, 0, 1);
    }
    const unsigned short* brow =
        frag + ((size_t)(tap * KB_) * NT_ + wave * 4) * 512 + lane * 16;

#pragma unroll
    for (int kb = 0; kb < KB_; ++kb) {
      union { v4u u[2]; v16bf v; } a0, a1;
      const unsigned short* ap0 = arow0 + kb * 32 + khalf8;
      const unsigned short* ap1 = arow1 + kb * 32 + khalf8;
      a0.u[0] = *(const v4u*)(ap0);        // k_local = khalf*8 + 0..7
      a0.u[1] = *(const v4u*)(ap0 + 16);   // k_local = 16 + khalf*8 + 0..7
      a1.u[0] = *(const v4u*)(ap1);
      a1.u[1] = *(const v4u*)(ap1 + 16);
#pragma unroll
      for (int t = 0; t < 4; ++t) {
        union { v4u u[2]; v16bf v; } bb;
        const unsigned short* bp = brow + (size_t)kb * NT_ * 512 + t * 512;
        bb.u[0] = *(const v4u*)(bp);
        bb.u[1] = *(const v4u*)(bp + 8);
        acc[0][t] = __builtin_amdgcn_wmma_f32_16x16x32_bf16(
            false, a0.v, false, bb.v, (short)0, acc[0][t], false, false);
        acc[1][t] = __builtin_amdgcn_wmma_f32_16x16x32_bf16(
            false, a1.v, false, bb.v, (short)0, acc[1][t], false, false);
      }
    }
  }

  // D layout: VGPR r, lane l -> row = r + (l<16 ? 0 : 8), col = (l&15)
  // Flattened pixel index == output row index, so offset = m*F + n directly.
  const int rowoff = (lane >> 4) * 8;
  const int ncol0  = wave * 64 + (lane & 15);
#pragma unroll
  for (int mt = 0; mt < 2; ++mt) {
#pragma unroll
    for (int r = 0; r < 8; ++r) {
      float* o = out + (size_t)(mbase + mt * 16 + rowoff + r) * F_ + ncol0;
      o[0]  = acc[mt][0][r];
      o[16] = acc[mt][1][r];
      o[32] = acc[mt][2][r];
      o[48] = acc[mt][3][r];
    }
  }
}

// ---------------------------------------------------------------------------
extern "C" void kernel_launch(void* const* d_in, const int* in_sizes, int n_in,
                              void* d_out, int out_size, void* d_ws, size_t ws_size,
                              hipStream_t stream) {
  const float* inputs = (const float*)d_in[0];
  const float* P      = (const float*)d_in[1];
  const float* Q      = (const float*)d_in[2];
  const float* S      = (const float*)d_in[3];
  float* out          = (float*)d_out;

  char* ws = (char*)d_ws;
  unsigned short* inbf = (unsigned short*)(ws + WS_INBF_OFF);
  float*          w1   = (float*)(ws + WS_W1_OFF);
  unsigned short* frag = (unsigned short*)(ws + WS_FRAG_OFF);

  {
    int n = B_ * H_ * W_ * C_;
    cvt_bf16_kernel<<<(n + 511) / 512, 512, 0, stream>>>(inputs, inbf, n);
  }
  {
    int n = TAPS_ * C_ * F_;
    w1_kernel<<<(n + 255) / 256, 256, 0, stream>>>(Q, S, w1);
  }
  {
    int n = TAPS_ * KB_ * NT_ * 32 * 16;
    frag_kernel<<<(n + 255) / 256, 256, 0, stream>>>(P, w1, frag);
  }
  conv_wmma_kernel<<<M_ / 32, 128, 0, stream>>>(inbf, frag, out);
}
